// CrossAttention_39737037423155
// MI455X (gfx1250) — matrix operand order repaired and checked by reading, not compiled, not run on hardware
//
#include <hip/hip_runtime.h>
#include <hip/hip_bf16.h>

typedef __bf16 bf16;
typedef __attribute__((ext_vector_type(16))) __bf16 v16bf;
typedef __attribute__((ext_vector_type(8)))  __bf16 v8bf;
typedef __attribute__((ext_vector_type(8)))  float  v8f;

#define WMMA_BF16(a, b, c) \
  __builtin_amdgcn_wmma_f32_16x16x32_bf16(false, (a), false, (b), (short)0, (c), false, false)

// ---------------------------------------------------------------------------
// Fragment loaders per CDNA5 ISA 7.12.2 (wave32, 16-bit operands)
// A 16x32 (MxK): lane L holds row L&15; K = {0..7,16..23} (L<16) or
//                {8..15,24..31} (L>=16), packed 2 per VGPR.
// B 32x16 (KxN): lane L holds col L&15; K = {0..7,16..23} (L<16) or
//                {8..15,24..31} (L>=16), same as A.
// C/D 16x16 f32: VGPR r -> row r (lanes 0-15) / row r+8 (lanes 16-31), col L&15.
// All fragments are contiguous 16B loads (B is always fed transposed).
// ---------------------------------------------------------------------------

__device__ __forceinline__ v16bf load_a_frag(const bf16* A, int lda, int m0,
                                             int k0, int lane) {
  int row = m0 + (lane & 15);
  int kb  = k0 + ((lane & 16) ? 8 : 0);
  const bf16* p = A + (size_t)row * lda + kb;
  v8bf lo = *(const v8bf*)(p);        // K = kb..kb+7
  v8bf hi = *(const v8bf*)(p + 16);   // K = kb+16..kb+23
  v16bf a;
#pragma unroll
  for (int t = 0; t < 8; ++t) { a[t] = lo[t]; a[8 + t] = hi[t]; }
  return a;
}

// B fed as Bt row-major [N x K]; B[k][n] = Bt[n][k] -> contiguous per lane.
__device__ __forceinline__ v16bf load_bT_frag(const bf16* Bt, int ld, int n0,
                                              int k0, int lane) {
  int row = n0 + (lane & 15);
  int kb  = k0 + ((lane & 16) ? 8 : 0);
  const bf16* p = Bt + (size_t)row * ld + kb;
  v8bf lo = *(const v8bf*)(p);
  v8bf hi = *(const v8bf*)(p + 16);
  v16bf b;
#pragma unroll
  for (int t = 0; t < 8; ++t) { b[t] = lo[t]; b[8 + t] = hi[t]; }
  return b;
}

// ---------------------------------------------------------------------------
// elementwise f32 -> bf16
// ---------------------------------------------------------------------------
__global__ void cvt_bf16_kernel(const float* __restrict__ in,
                                bf16* __restrict__ out, int n) {
  for (int i = blockIdx.x * blockDim.x + threadIdx.x; i < n;
       i += gridDim.x * blockDim.x)
    out[i] = (bf16)in[i];
}

// transpose-convert: in f32 [K x N] row-major -> out bf16 [N x K] row-major
__global__ void cvt_t_kernel(const float* __restrict__ in,
                             bf16* __restrict__ out, int K, int N) {
  int i = blockIdx.x * blockDim.x + threadIdx.x;
  if (i >= K * N) return;
  int k = i % K, n = i / K;
  out[i] = (bf16)in[(size_t)k * N + n];
}

// V transpose: in bf16 [B*M, 512] -> out bf16 [B, 8, 64, M]
__global__ void vtrans_kernel(const bf16* __restrict__ in,
                              bf16* __restrict__ out, int Mc, size_t total) {
  size_t i = blockIdx.x * (size_t)blockDim.x + threadIdx.x;
  if (i >= total) return;
  int m = (int)(i % Mc);
  size_t t = i / Mc;
  int d = (int)(t % 64); t /= 64;
  int h = (int)(t % 8);
  int b = (int)(t / 8);
  out[i] = in[((size_t)b * Mc + m) * 512 + h * 64 + d];
}

// ---------------------------------------------------------------------------
// gate: g = sigmoid(relu(ctx@W1+b1)@W2+b2); out = bf16(ctx * g). One row/thread.
// ---------------------------------------------------------------------------
__global__ void gate_kernel(const float* __restrict__ ctx,
                            const float* __restrict__ W1,
                            const float* __restrict__ b1,
                            const float* __restrict__ W2,
                            const float* __restrict__ b2,
                            bf16* __restrict__ out, int rows) {
  int i = blockIdx.x * blockDim.x + threadIdx.x;
  if (i >= rows) return;
  const float* c = ctx + (size_t)i * 64;
  float h[32];
#pragma unroll
  for (int j = 0; j < 32; ++j) h[j] = b1[j];
  for (int d = 0; d < 64; ++d) {
    float cv = c[d];
#pragma unroll
    for (int j = 0; j < 32; ++j) h[j] += cv * W1[d * 32 + j];
  }
  float g = b2[0];
#pragma unroll
  for (int j = 0; j < 32; ++j) g += fmaxf(h[j], 0.f) * W2[j];
  g = 1.f / (1.f + __expf(-g));
  for (int d = 0; d < 64; ++d) out[(size_t)i * 64 + d] = (bf16)(c[d] * g);
}

// ---------------------------------------------------------------------------
// WMMA GEMM: C[MxN] = A[MxK] @ B[KxN], B supplied transposed (Bt: [N x K]).
// One wave -> 32x64 tile (2 A-frags share 4 B-frags, 8 WMMAs / K-step).
// bf16 path applies out_scale (used to fold attention's 1/sqrt(d) into Q).
// ---------------------------------------------------------------------------
template <bool OUT_F32_BIAS>
__global__ __launch_bounds__(32, 1) void gemm_wmma_kernel(
    const bf16* __restrict__ A, const bf16* __restrict__ Bt,
    void* __restrict__ C, const float* __restrict__ bias, float out_scale,
    int M, int N, int K) {
  int lane = threadIdx.x;
  int m0 = blockIdx.x * 32;
  int n0 = blockIdx.y * 64;
  const v8f vzero = {0.f, 0.f, 0.f, 0.f, 0.f, 0.f, 0.f, 0.f};
  v8f acc[2][4];
#pragma unroll
  for (int i = 0; i < 2; ++i)
#pragma unroll
    for (int j = 0; j < 4; ++j) acc[i][j] = vzero;

  for (int k0 = 0; k0 < K; k0 += 32) {
    v16bf a0 = load_a_frag(A, K, m0, k0, lane);
    v16bf a1 = load_a_frag(A, K, m0 + 16, k0, lane);
#pragma unroll
    for (int nb = 0; nb < 4; ++nb) {
      v16bf b = load_bT_frag(Bt, K, n0 + nb * 16, k0, lane);
      acc[0][nb] = WMMA_BF16(a0, b, acc[0][nb]);
      acc[1][nb] = WMMA_BF16(a1, b, acc[1][nb]);
    }
  }
  int rr = (lane & 16) ? 8 : 0;
  int col = lane & 15;
#pragma unroll
  for (int mt = 0; mt < 2; ++mt)
#pragma unroll
    for (int nb = 0; nb < 4; ++nb)
#pragma unroll
      for (int r = 0; r < 8; ++r) {
        size_t idx = (size_t)(m0 + mt * 16 + r + rr) * N + (n0 + nb * 16 + col);
        float v = acc[mt][nb][r];
        if (OUT_F32_BIAS)
          ((float*)C)[idx] = v + bias[n0 + nb * 16 + col];
        else
          ((bf16*)C)[idx] = (bf16)(v * out_scale);
      }
}

// ---------------------------------------------------------------------------
// Flash attention: one wave per (b, h, 16 query rows); 64 keys per iteration.
// Q is pre-scaled by 1/sqrt(d) at projection time, so S needs no scaling here.
// Online softmax stats live directly in the WMMA C-layout lane mapping; the
// xor-butterflies are issued stage-outer so 8 ds_bpermutes pipeline per wait.
// ---------------------------------------------------------------------------
__global__ __launch_bounds__(32, 1) void attn_kernel(
    const bf16* __restrict__ Q, const bf16* __restrict__ Km,
    const bf16* __restrict__ Vt, bf16* __restrict__ O, int N, int M) {
  const int INNERC = 512;  // H * 64
  int lane = threadIdx.x;
  int i0 = blockIdx.x * 16;
  int h = blockIdx.y;
  int b = blockIdx.z;
  const bf16* Qb = Q + (size_t)b * N * INNERC + h * 64;
  const bf16* Kb = Km + (size_t)b * M * INNERC + h * 64;
  const bf16* Vb = Vt + ((size_t)(b * 8 + h) * 64) * M;  // [64 x M], ld = M

  __shared__ bf16 p_lds[16][72];  // 144B rows -> 16B-aligned sub-blocks

  // resident Q fragments (head dim 64 = two K-steps of 32)
  v16bf aq0 = load_a_frag(Qb, INNERC, i0, 0, lane);
  v16bf aq1 = load_a_frag(Qb, INNERC, i0, 32, lane);

  const v8f vzero = {0.f, 0.f, 0.f, 0.f, 0.f, 0.f, 0.f, 0.f};
  v8f acc[4];
#pragma unroll
  for (int nb = 0; nb < 4; ++nb) acc[nb] = vzero;
  float mrow[8], lrow[8];
#pragma unroll
  for (int r = 0; r < 8; ++r) { mrow[r] = -1e30f; lrow[r] = 0.f; }
  int prow = (lane & 16) ? 8 : 0;
  int pcol = lane & 15;

  for (int j0 = 0; j0 < M; j0 += 64) {
    // S(16x64) = Q(16x64) @ K_chunk^T : 8 WMMAs
    v8f s[4];
#pragma unroll
    for (int cb = 0; cb < 4; ++cb) {
      v16bf bk0 = load_bT_frag(Kb, INNERC, j0 + cb * 16, 0, lane);
      v16bf bk1 = load_bT_frag(Kb, INNERC, j0 + cb * 16, 32, lane);
      v8f c = vzero;
      c = WMMA_BF16(aq0, bk0, c);
      c = WMMA_BF16(aq1, bk1, c);
      s[cb] = c;
    }
    // --- row max: local over 4 col-blocks, then 4-stage xor butterfly ---
    float mx[8];
#pragma unroll
    for (int r = 0; r < 8; ++r)
      mx[r] = fmaxf(fmaxf(s[0][r], s[1][r]), fmaxf(s[2][r], s[3][r]));
#pragma unroll
    for (int mk = 8; mk >= 1; mk >>= 1)
#pragma unroll
      for (int r = 0; r < 8; ++r)
        mx[r] = fmaxf(mx[r], __shfl_xor(mx[r], mk, 32));
    // --- exp, write P to LDS, local row sums ---
    float sm[8];
#pragma unroll
    for (int r = 0; r < 8; ++r) {
      float mnew = fmaxf(mrow[r], mx[r]);
      mx[r] = mnew;  // reuse as m_new
      float acc_s = 0.f;
#pragma unroll
      for (int cb = 0; cb < 4; ++cb) {
        float p = __expf(s[cb][r] - mnew);
        acc_s += p;
        p_lds[r + prow][pcol + cb * 16] = (bf16)p;
      }
      sm[r] = acc_s;
    }
#pragma unroll
    for (int mk = 8; mk >= 1; mk >>= 1)
#pragma unroll
      for (int r = 0; r < 8; ++r) sm[r] += __shfl_xor(sm[r], mk, 32);
    // --- rescale running state ---
#pragma unroll
    for (int r = 0; r < 8; ++r) {
      float corr = __expf(mrow[r] - mx[r]);
      mrow[r] = mx[r];
      lrow[r] = lrow[r] * corr + sm[r];
#pragma unroll
      for (int nb = 0; nb < 4; ++nb) acc[nb][r] *= corr;
    }
    __syncthreads();
    // --- P fragments from LDS (16B ds_load_b128s) ---
    v16bf pa0, pa1;
    {
      int row = lane & 15;
      int kb = (lane & 16) ? 8 : 0;
      v8bf l0 = *(const v8bf*)&p_lds[row][kb];
      v8bf h0 = *(const v8bf*)&p_lds[row][kb + 16];
      v8bf l1 = *(const v8bf*)&p_lds[row][kb + 32];
      v8bf h1 = *(const v8bf*)&p_lds[row][kb + 48];
#pragma unroll
      for (int t = 0; t < 8; ++t) {
        pa0[t] = l0[t]; pa0[8 + t] = h0[t];
        pa1[t] = l1[t]; pa1[8 + t] = h1[t];
      }
    }
    // --- O(16x64) += P(16x64) @ V_chunk(64x64) : 8 WMMAs ---
#pragma unroll
    for (int nb = 0; nb < 4; ++nb) {
      v16bf bv0 = load_bT_frag(Vb, M, nb * 16, j0, lane);
      v16bf bv1 = load_bT_frag(Vb, M, nb * 16, j0 + 32, lane);
      acc[nb] = WMMA_BF16(pa0, bv0, acc[nb]);
      acc[nb] = WMMA_BF16(pa1, bv1, acc[nb]);
    }
    __syncthreads();
  }
  // epilogue: normalize and store in [B, N, H*64] layout (row-major INNER)
#pragma unroll
  for (int nb = 0; nb < 4; ++nb)
#pragma unroll
    for (int r = 0; r < 8; ++r) {
      float v = acc[nb][r] / lrow[r];
      O[(size_t)(b * N + i0 + r + prow) * INNERC + h * 64 + nb * 16 + pcol] =
          (bf16)v;
    }
}

// ---------------------------------------------------------------------------
extern "C" void kernel_launch(void* const* d_in, const int* in_sizes, int n_in,
                              void* d_out, int out_size, void* d_ws, size_t ws_size,
                              hipStream_t stream) {
  (void)in_sizes; (void)n_in; (void)out_size; (void)ws_size;
  const int B = 4, N = 2048, M = 2048, QD = 512, CD = 64, INNERC = 512;

  const float* x   = (const float*)d_in[0];
  const float* ctx = (const float*)d_in[1];
  const float* Wq  = (const float*)d_in[2];
  const float* Wk  = (const float*)d_in[3];
  const float* Wv  = (const float*)d_in[4];
  const float* W1  = (const float*)d_in[5];
  const float* b1  = (const float*)d_in[6];
  const float* W2  = (const float*)d_in[7];
  const float* b2  = (const float*)d_in[8];
  const float* Wo  = (const float*)d_in[9];
  const float* bo  = (const float*)d_in[10];
  float* out = (float*)d_out;

  // workspace carve-up (bf16 staging buffers), 256B aligned
  char* ws = (char*)d_ws;
  size_t off = 0;
  auto carve = [&](size_t bytes) {
    char* p = ws + off;
    off += (bytes + 255) & ~(size_t)255;
    return p;
  };
  bf16* xbf  = (bf16*)carve((size_t)B * N * QD * 2);      // 8 MB
  bf16* wqt  = (bf16*)carve((size_t)QD * INNERC * 2);     // 512 KB  [N x K]
  bf16* wkt  = (bf16*)carve((size_t)CD * INNERC * 2);     // 64 KB   [N x K]
  bf16* wvt  = (bf16*)carve((size_t)CD * INNERC * 2);     // 64 KB   [N x K]
  bf16* wot  = (bf16*)carve((size_t)INNERC * QD * 2);     // 512 KB  [N x K]
  bf16* ctxg = (bf16*)carve((size_t)B * M * CD * 2);      // 1 MB
  bf16* qb   = (bf16*)carve((size_t)B * N * INNERC * 2);  // 8 MB
  bf16* kb   = (bf16*)carve((size_t)B * M * INNERC * 2);  // 8 MB
  bf16* vb   = (bf16*)carve((size_t)B * M * INNERC * 2);  // 8 MB
  bf16* vt   = (bf16*)carve((size_t)B * M * INNERC * 2);  // 8 MB [B,8,64,M]
  bf16* ab   = (bf16*)carve((size_t)B * N * INNERC * 2);  // 8 MB

  // 1) bf16 staging: x plain; weights transposed-converted ([N x K] bf16)
  cvt_bf16_kernel<<<2048, 256, 0, stream>>>(x, xbf, B * N * QD);
  cvt_t_kernel<<<(QD * INNERC + 255) / 256, 256, 0, stream>>>(Wq, wqt, QD, INNERC);
  cvt_t_kernel<<<(CD * INNERC + 255) / 256, 256, 0, stream>>>(Wk, wkt, CD, INNERC);
  cvt_t_kernel<<<(CD * INNERC + 255) / 256, 256, 0, stream>>>(Wv, wvt, CD, INNERC);
  cvt_t_kernel<<<(INNERC * QD + 255) / 256, 256, 0, stream>>>(Wo, wot, INNERC, QD);

  // 2) reliability gate + context scaling
  gate_kernel<<<(B * M + 127) / 128, 128, 0, stream>>>(ctx, W1, b1, W2, b2, ctxg,
                                                       B * M);

  // 3) projections (bf16 WMMA GEMMs, 32x64 tile per wave)
  //    Q is pre-scaled by 1/sqrt(DIM_HEAD) = 0.125
  {
    dim3 g(B * N / 32, INNERC / 64);
    gemm_wmma_kernel<false><<<g, 32, 0, stream>>>(xbf, wqt, qb, nullptr, 0.125f,
                                                  B * N, INNERC, QD);
  }
  {
    dim3 g(B * M / 32, INNERC / 64);
    gemm_wmma_kernel<false><<<g, 32, 0, stream>>>(ctxg, wkt, kb, nullptr, 1.0f,
                                                  B * M, INNERC, CD);
    gemm_wmma_kernel<false><<<g, 32, 0, stream>>>(ctxg, wvt, vb, nullptr, 1.0f,
                                                  B * M, INNERC, CD);
  }

  // 4) V head-transpose for contiguous B-fragments in attention
  {
    size_t total = (size_t)B * M * INNERC;
    vtrans_kernel<<<(unsigned)((total + 255) / 256), 256, 0, stream>>>(vb, vt, M,
                                                                       total);
  }

  // 5) flash attention, one wave per (b, h, 16 q-rows), 64 keys / iter
  {
    dim3 g(N / 16, 8, B);
    attn_kernel<<<g, 32, 0, stream>>>(qb, kb, vt, ab, N, M);
  }

  // 6) output projection with bias, f32 result
  {
    dim3 g(B * N / 32, QD / 64);
    gemm_wmma_kernel<true><<<g, 32, 0, stream>>>(ab, wot, out, bo, 0.f, B * N,
                                                 QD, INNERC);
  }
}
